// BiEncoderModel_13520557048072
// MI455X (gfx1250) — matrix operand order repaired and checked by reading, not compiled, not run on hardware
//
#include <hip/hip_runtime.h>

#define NN 2048
#define HD 768
#define LW 30
#define NH 6

typedef __attribute__((ext_vector_type(16))) __bf16 v16bf;
typedef __attribute__((ext_vector_type(8)))  float  v8f;

union ABfrag {
    v16bf v;
    uint4 q[2];
};

__device__ __forceinline__ unsigned short f2bf(float f) {
    unsigned int u = __float_as_uint(f);
    u += 0x7FFFu + ((u >> 16) & 1u);     // round-to-nearest-even
    return (unsigned short)(u >> 16);
}

__device__ __forceinline__ v8f zero8() {
    v8f z = {0.f, 0.f, 0.f, 0.f, 0.f, 0.f, 0.f, 0.f};
    return z;
}

__device__ __forceinline__ float wave_sum(float v) {
#pragma unroll
    for (int o = 16; o; o >>= 1) v += __shfl_xor(v, o, 32);
    return v;
}
__device__ __forceinline__ float wave_max(float v) {
#pragma unroll
    for (int o = 16; o; o >>= 1) v = fmaxf(v, __shfl_xor(v, o, 32));
    return v;
}

// A-matrix 16x32 bf16 fragment from LDS (row-major, stride rs elements).
// lanes 0-15: row=row0+lane, K = k0+[0..7], k0+[16..23]
// lanes 16-31: row=row0+lane-16, K = k0+[8..15], k0+[24..31]
__device__ __forceinline__ v16bf load_a_frag_lds(const unsigned short* base, int rs,
                                                 int row0, int k0) {
    int ln = threadIdx.x & 31;
    int row = row0 + (ln & 15);
    int kb  = k0 + ((ln >> 4) << 3);
    const unsigned short* p = base + row * rs + kb;
    ABfrag f;
    f.q[0] = *(const uint4*)p;
    f.q[1] = *(const uint4*)(p + 16);
    return f.v;
}

// B-matrix 32x16 bf16 fragment where B = src^T, src row-major [n][k] (ld elements).
// lane holds column n0+(lane&15), K = k0 + (lane<16 ? [0..7],[16..23] : [8..15],[24..31])
__device__ __forceinline__ v16bf load_b_frag_rm(const unsigned short* src, int ld,
                                                int n0, int k0) {
    int ln = threadIdx.x & 31;
    const unsigned short* p = src + (size_t)(n0 + (ln & 15)) * ld + (k0 + ((ln >> 4) << 3));
    ABfrag f;
    f.q[0] = *(const uint4*)p;
    f.q[1] = *(const uint4*)(p + 16);
    return f.v;
}

__device__ __forceinline__ v8f wmma_bf16(v16bf a, v16bf b, v8f c) {
    return __builtin_amdgcn_wmma_f32_16x16x32_bf16(false, a, false, b, (short)0, c,
                                                   false, false);
}

// ---------------- kernel 1: convert W -> bf16; aspect_v -> transposed bf16 ----------------
__global__ void k_convert(const float* __restrict__ W, const float* __restrict__ aspect,
                          unsigned short* __restrict__ Wb, unsigned short* __restrict__ Vt) {
    int tid = blockIdx.x * blockDim.x + threadIdx.x;
    const int WNel = NH * HD * HD;
    if (tid < WNel) {
        Wb[tid] = f2bf(W[tid]);
    } else {
        int idx = tid - WNel;
        if (idx < NN * HD) {
            int n = idx / HD, e = idx % HD;
            Vt[(size_t)e * NN + n] = f2bf(aspect[idx]);
        }
    }
}

// ---------------- kernel 2: Fh = feature@W^T + b, normalize rows, mean over L -> M ----------
#define A_RS 784   // 768+16 ushort, 16B-aligned stride
#define F_RS 772   // 768+4 float, 16B-aligned stride

__global__ __launch_bounds__(256) void k_featM(const float* __restrict__ feature,
                                               const unsigned short* __restrict__ Wb,
                                               const float* __restrict__ bias,
                                               unsigned short* __restrict__ QM) {
    extern __shared__ __align__(16) char smem[];
    unsigned short* Al = (unsigned short*)smem;                        // 32 x 784 bf16
    float* Fl = (float*)(smem + 32 * A_RS * 2);                        // 32 x 772 f32
    float* rn = (float*)(smem + 32 * A_RS * 2 + 32 * F_RS * 4);        // 32 f32

    const int n = blockIdx.x;
    const int tid = threadIdx.x;
    const int w = tid >> 5, ln = tid & 31;
    const int rowoff = (ln >> 4) ? 8 : 0;

    // stage feature[n] (30x768 f32) as bf16 into LDS; pad rows 30,31 with zero
    const float* fp = feature + (size_t)n * (LW * HD);
    for (int i = tid; i < LW * (HD / 4); i += 256) {
        int l = i / (HD / 4), c4 = i % (HD / 4);
        float4 v = ((const float4*)fp)[i];
        unsigned short* dst = Al + l * A_RS + c4 * 4;
        dst[0] = f2bf(v.x); dst[1] = f2bf(v.y); dst[2] = f2bf(v.z); dst[3] = f2bf(v.w);
    }
    for (int i = tid; i < 2 * HD; i += 256) {
        Al[(LW + i / HD) * A_RS + (i % HD)] = 0;
    }
    __syncthreads();

    for (int h = 0; h < NH; ++h) {
        const unsigned short* Wh = Wb + (size_t)h * HD * HD;
        v8f acc[2][6];
#pragma unroll
        for (int r = 0; r < 2; ++r)
#pragma unroll
            for (int j = 0; j < 6; ++j) acc[r][j] = zero8();

        for (int k = 0; k < HD; k += 32) {
            v16bf a0 = load_a_frag_lds(Al, A_RS, 0, k);
            v16bf a1 = load_a_frag_lds(Al, A_RS, 16, k);
#pragma unroll
            for (int j = 0; j < 6; ++j) {
                int c = w * 6 + j;
                v16bf bf = load_b_frag_rm(Wh, HD, c * 16, k);
                acc[0][j] = wmma_bf16(a0, bf, acc[0][j]);
                acc[1][j] = wmma_bf16(a1, bf, acc[1][j]);
            }
        }
        // write Fh (+bias) to LDS
#pragma unroll
        for (int j = 0; j < 6; ++j) {
            int col = (w * 6 + j) * 16 + (ln & 15);
            float bv = bias[h * HD + col];
#pragma unroll
            for (int v = 0; v < 8; ++v) {
                Fl[(v + rowoff) * F_RS + col]      = acc[0][j][v] + bv;
                Fl[(16 + v + rowoff) * F_RS + col] = acc[1][j][v] + bv;
            }
        }
        __syncthreads();
        // per-row reciprocal norms (rows 0..29)
        for (int r = w; r < 32; r += 8) {
            if (r < LW) {
                float s = 0.f;
                for (int c = ln; c < HD; c += 32) {
                    float x = Fl[r * F_RS + c];
                    s += x * x;
                }
                s = wave_sum(s);
                if (ln == 0) rn[r] = 1.0f / fmaxf(sqrtf(s), 1e-12f);
            }
        }
        __syncthreads();
        // mean over L of normalized rows -> M, store into QM[h][n][768+e]
        for (int e = tid; e < HD; e += 256) {
            float s = 0.f;
            for (int l = 0; l < LW; ++l) s += Fl[l * F_RS + e] * rn[l];
            s *= (1.0f / LW);
            QM[((size_t)h * NN + n) * 1536 + HD + e] = f2bf(s);
        }
        __syncthreads();
    }
}

// ---------------- kernel 3: Q = aspect@W^T + b, normalize rows -> QM[...][0:768] -----------
__global__ __launch_bounds__(256) void k_aspectQ(const float* __restrict__ aspect,
                                                 const unsigned short* __restrict__ Wb,
                                                 const float* __restrict__ bias,
                                                 unsigned short* __restrict__ QM) {
    extern __shared__ __align__(16) char smem[];
    unsigned short* Al = (unsigned short*)smem;
    float* Fl = (float*)(smem + 32 * A_RS * 2);
    float* rn = (float*)(smem + 32 * A_RS * 2 + 32 * F_RS * 4);

    const int n0 = blockIdx.x * 32;
    const int tid = threadIdx.x;
    const int w = tid >> 5, ln = tid & 31;
    const int rowoff = (ln >> 4) ? 8 : 0;

    const float* ap = aspect + (size_t)n0 * HD;
    for (int i = tid; i < 32 * (HD / 4); i += 256) {
        int r = i / (HD / 4), c4 = i % (HD / 4);
        float4 v = ((const float4*)ap)[i];
        unsigned short* dst = Al + r * A_RS + c4 * 4;
        dst[0] = f2bf(v.x); dst[1] = f2bf(v.y); dst[2] = f2bf(v.z); dst[3] = f2bf(v.w);
    }
    __syncthreads();

    for (int h = 0; h < NH; ++h) {
        const unsigned short* Wh = Wb + (size_t)h * HD * HD;
        v8f acc[2][6];
#pragma unroll
        for (int r = 0; r < 2; ++r)
#pragma unroll
            for (int j = 0; j < 6; ++j) acc[r][j] = zero8();

        for (int k = 0; k < HD; k += 32) {
            v16bf a0 = load_a_frag_lds(Al, A_RS, 0, k);
            v16bf a1 = load_a_frag_lds(Al, A_RS, 16, k);
#pragma unroll
            for (int j = 0; j < 6; ++j) {
                int c = w * 6 + j;
                v16bf bf = load_b_frag_rm(Wh, HD, c * 16, k);
                acc[0][j] = wmma_bf16(a0, bf, acc[0][j]);
                acc[1][j] = wmma_bf16(a1, bf, acc[1][j]);
            }
        }
#pragma unroll
        for (int j = 0; j < 6; ++j) {
            int col = (w * 6 + j) * 16 + (ln & 15);
            float bv = bias[h * HD + col];
#pragma unroll
            for (int v = 0; v < 8; ++v) {
                Fl[(v + rowoff) * F_RS + col]      = acc[0][j][v] + bv;
                Fl[(16 + v + rowoff) * F_RS + col] = acc[1][j][v] + bv;
            }
        }
        __syncthreads();
        for (int r = w; r < 32; r += 8) {
            float s = 0.f;
            for (int c = ln; c < HD; c += 32) {
                float x = Fl[r * F_RS + c];
                s += x * x;
            }
            s = wave_sum(s);
            if (ln == 0) rn[r] = 1.0f / fmaxf(sqrtf(s), 1e-12f);
        }
        __syncthreads();
        for (int e = tid; e < HD; e += 256) {
            for (int r = 0; r < 32; ++r) {
                float q = Fl[r * F_RS + e] * rn[r];
                QM[((size_t)h * NN + n0 + r) * 1536 + e] = f2bf(q);
            }
        }
        __syncthreads();
    }
}

// ---------------- kernel 4: S = (QM QM^T)*dmask, ==0 -> -1e30, softmax, @aspect_v ----------
#define AO_RS 1552   // 1536+16 ushort
#define S_RS  2052   // 2048+4 float
#define P_RS  2064   // 2048+16 ushort

__global__ __launch_bounds__(256) void k_attn(const unsigned short* __restrict__ QM,
                                              const float* __restrict__ dmask,
                                              const unsigned short* __restrict__ Vt,
                                              float* __restrict__ outP) {
    extern __shared__ __align__(16) char smem[];
    unsigned short* Ao = (unsigned short*)smem;                               // 16 x 1552 bf16
    float* Sl = (float*)(smem + 16 * AO_RS * 2);                              // 16 x 2052 f32
    unsigned short* Pl = (unsigned short*)(smem + 16 * AO_RS * 2 + 16 * S_RS * 4); // 16x2064

    const int n0 = blockIdx.x * 16;
    const int h = blockIdx.y;
    const int tid = threadIdx.x;
    const int w = tid >> 5, ln = tid & 31;
    const int rowoff = (ln >> 4) ? 8 : 0;
    const unsigned short* QMh = QM + (size_t)h * NN * 1536;

    // stage own 16 rows of [Q|M] (16 x 1536 bf16)
    for (int i = tid; i < 16 * (1536 / 8); i += 256) {
        int r = i / 192, c = i % 192;
        ((uint4*)(Ao + r * AO_RS))[c] = ((const uint4*)(QMh + (size_t)(n0 + r) * 1536))[c];
    }
    __syncthreads();

    // S-GEMM: 128 m-tiles, wave w owns tiles [16w, 16w+16), two register groups of 8
    for (int g = 0; g < 2; ++g) {
        v8f acc[8];
#pragma unroll
        for (int j = 0; j < 8; ++j) acc[j] = zero8();
        for (int k = 0; k < 1536; k += 32) {
            v16bf a = load_a_frag_lds(Ao, AO_RS, 0, k);
#pragma unroll
            for (int j = 0; j < 8; ++j) {
                int c = w * 16 + g * 8 + j;
                v16bf bf = load_b_frag_rm(QMh, 1536, c * 16, k);
                acc[j] = wmma_bf16(a, bf, acc[j]);
            }
        }
#pragma unroll
        for (int j = 0; j < 8; ++j) {
            int m = (w * 16 + g * 8 + j) * 16 + (ln & 15);
#pragma unroll
            for (int v = 0; v < 8; ++v) {
                int row = v + rowoff;
                float dm = dmask[(size_t)(n0 + row) * NN + m];
                float val = acc[j][v] * dm;
                if (val == 0.0f) val = -1e30f;
                Sl[row * S_RS + m] = val;
            }
        }
    }
    __syncthreads();

    // row softmax (each wave owns 2 rows), emit bf16 probabilities
    for (int r = w; r < 16; r += 8) {
        float mx = -3.0e38f;
        for (int c = ln; c < NN; c += 32) mx = fmaxf(mx, Sl[r * S_RS + c]);
        mx = wave_max(mx);
        float sum = 0.f;
        for (int c = ln; c < NN; c += 32) {
            float e = __expf(Sl[r * S_RS + c] - mx);
            Sl[r * S_RS + c] = e;
            sum += e;
        }
        sum = wave_sum(sum);
        float inv = 1.0f / sum;
        for (int c = ln; c < NN; c += 32) Pl[r * P_RS + c] = f2bf(Sl[r * S_RS + c] * inv);
    }
    __syncthreads();

    // P @ V: 48 e-tiles, wave w owns 6; B from pre-transposed Vt[e][m]
    v8f acc2[6];
#pragma unroll
    for (int j = 0; j < 6; ++j) acc2[j] = zero8();
    for (int k = 0; k < NN; k += 32) {
        v16bf a = load_a_frag_lds(Pl, P_RS, 0, k);
#pragma unroll
        for (int j = 0; j < 6; ++j) {
            v16bf bf = load_b_frag_rm(Vt, NN, (w * 6 + j) * 16, k);
            acc2[j] = wmma_bf16(a, bf, acc2[j]);
        }
    }
    float* op = outP + ((size_t)h * NN + n0) * HD;
#pragma unroll
    for (int j = 0; j < 6; ++j) {
        int col = (w * 6 + j) * 16 + (ln & 15);
#pragma unroll
        for (int v = 0; v < 8; ++v) op[(size_t)(v + rowoff) * HD + col] = acc2[j][v];
    }
}

// ---------------- kernel 5: mean over heads --------------------------------------------------
__global__ void k_reduce(const float* __restrict__ outP, float* __restrict__ out) {
    int i = blockIdx.x * blockDim.x + threadIdx.x;
    if (i < NN * HD) {
        float s = 0.f;
#pragma unroll
        for (int h = 0; h < NH; ++h) s += outP[(size_t)h * NN * HD + i];
        out[i] = s * (1.0f / NH);
    }
}

extern "C" void kernel_launch(void* const* d_in, const int* in_sizes, int n_in,
                              void* d_out, int out_size, void* d_ws, size_t ws_size,
                              hipStream_t stream) {
    (void)in_sizes; (void)n_in; (void)out_size; (void)ws_size;
    const float* feature = (const float*)d_in[0];
    const float* aspect  = (const float*)d_in[1];
    const float* dmask   = (const float*)d_in[2];
    const float* W       = (const float*)d_in[3];
    const float* bias    = (const float*)d_in[4];
    float* out = (float*)d_out;

    char* ws = (char*)d_ws;
    size_t off = 0;
    auto take = [&](size_t bytes) -> char* {
        char* p = ws + off;
        off += (bytes + 255) & ~(size_t)255;
        return p;
    };
    unsigned short* Wb   = (unsigned short*)take((size_t)NH * HD * HD * 2);   //  ~7 MB
    unsigned short* QM   = (unsigned short*)take((size_t)NH * NN * 1536 * 2); // ~38 MB
    unsigned short* Vt   = (unsigned short*)take((size_t)HD * NN * 2);        //  ~3 MB
    float*          outP = (float*)take((size_t)NH * NN * HD * 4);            // ~38 MB

    {
        int total = NH * HD * HD + NN * HD;
        k_convert<<<(total + 255) / 256, 256, 0, stream>>>(W, aspect, Wb, Vt);
    }
    const size_t smem_fm = 32 * A_RS * 2 + 32 * F_RS * 4 + 32 * 4;            // ~149 KB
    k_featM<<<NN, 256, smem_fm, stream>>>(feature, Wb, bias, QM);
    k_aspectQ<<<NN / 32, 256, smem_fm, stream>>>(aspect, Wb, bias, QM);

    const size_t smem_at = (size_t)16 * AO_RS * 2 + (size_t)16 * S_RS * 4 +
                           (size_t)16 * P_RS * 2;                              // ~247 KB
    dim3 ga(NN / 16, NH);
    k_attn<<<ga, 256, smem_at, stream>>>(QM, dmask, Vt, outP);

    k_reduce<<<(NN * HD + 255) / 256, 256, 0, stream>>>(outP, out);
}
